// SelfAttentionEnhanced_4243427688506
// MI455X (gfx1250) — matrix operand order repaired and checked by reading, not compiled, not run on hardware
//
#include <hip/hip_runtime.h>
#include <hip/hip_bf16.h>

typedef __attribute__((ext_vector_type(16))) _Float16 v16h;
typedef __attribute__((ext_vector_type(8)))  _Float16 v8h;
typedef __attribute__((ext_vector_type(8)))  float    v8f;
typedef __attribute__((ext_vector_type(4)))  unsigned int v4u;
typedef __attribute__((ext_vector_type(8)))  int      v8i;
typedef __attribute__((ext_vector_type(4)))  int      v4i;

#define B_   2
#define L_   2048
#define C_   768
#define H_   12
#define D_   64
#define NQKV 2304
#define M_   4096            // B_*L_
#define MAXLOGSCALE 4.60517018598809f   // log(100)

// ---------------------------------------------------------------------------
// Fragment loaders (layouts per CDNA5 ISA 7.12.2, wave32)
// ---------------------------------------------------------------------------
__device__ __forceinline__ v16h load_frag_a(const _Float16* base, int ld,
                                            int m0, int k0, int lane) {
  const _Float16* p = base + (size_t)(m0 + (lane & 15)) * ld + k0 + ((lane >> 4) * 8);
  v8h lo = *(const v8h*)(p);
  v8h hi = *(const v8h*)(p + 16);
  return __builtin_shufflevector(lo, hi, 0,1,2,3,4,5,6,7,8,9,10,11,12,13,14,15);
}

// B-matrix fragment gathered from row-major W[n][k] (i.e. B[k][n] = W[n][k]).
__device__ __forceinline__ v16h load_frag_b(const _Float16* base, int ld,
                                            int n0, int k0, int lane) {
  const _Float16* p = base + (size_t)(n0 + (lane & 15)) * ld + k0 + ((lane >> 4) * 8);
  v8h lo = *(const v8h*)(p);
  v8h hi = *(const v8h*)(p + 16);
  return __builtin_shufflevector(lo, hi, 0,1,2,3,4,5,6,7,8,9,10,11,12,13,14,15);
}

__device__ __forceinline__ v8f wmma_f16(v16h a, v16h b, v8f c) {
  return __builtin_amdgcn_wmma_f32_16x16x32_f16(false, a, false, b,
                                                (short)0, c, false, false);
}

__device__ __forceinline__ unsigned int rfl(unsigned int x) {
  return (unsigned int)__builtin_amdgcn_readfirstlane((int)x);
}

// ---------------------------------------------------------------------------
// Tensor Data Mover: stage a 2-D f16 tile (tile_d1 rows x tile_d0 elems,
// row stride stride_elems) from global into LDS at byte offset lds_off.
// D# per ISA 8.3/8.4: count=1, type=2, data_size=1 (2B), wg_mask=0.
// tensor dims set large (no clipping; tiles are in-bounds by construction).
// Toolchain uses the 6-arg builtin: (v4u, v8i, v4i, v4i, v8i, i32 cpol).
// ---------------------------------------------------------------------------
__device__ __forceinline__ void tdm_load_tile(unsigned int lds_off, const void* gptr,
                                              unsigned int tile_d0, unsigned int tile_d1,
                                              unsigned long long stride_elems) {
  unsigned long long ga = (unsigned long long)gptr;
  v4u g0;
  g0[0] = 1u;                                       // count=1, user descriptor
  g0[1] = rfl(lds_off);                             // lds_addr
  g0[2] = rfl((unsigned int)ga);                    // global_addr[31:0]
  g0[3] = (rfl((unsigned int)(ga >> 32)) & 0x01FFFFFFu) | 0x80000000u; // ga[56:32]|type=2
  v8i g1;
  g1[0] = (int)(1u << 16);                          // wg_mask=0, data_size=2B
  g1[1] = (int)(0xFFFFu << 16);                     // tensor_dim0[15:0]=0xFFFF
  g1[2] = (int)(0x00FFu | (0xFFFFu << 16));         // tensor_dim0 hi | tensor_dim1 lo
  g1[3] = (int)(0x00FFu | (tile_d0 << 16));         // tensor_dim1 hi | tile_dim0
  g1[4] = (int)(tile_d1 & 0xFFFFu);                 // tile_dim1, tile_dim2=0
  g1[5] = (int)rfl((unsigned int)stride_elems);     // tensor_dim0_stride[31:0]
  g1[6] = (int)((unsigned int)(stride_elems >> 32) & 0xFFFFu); // stride[47:32]
  g1[7] = 0;
  v4i z4 = {0, 0, 0, 0};
  v8i z8 = {0, 0, 0, 0, 0, 0, 0, 0};
  __builtin_amdgcn_tensor_load_to_lds(g0, g1, z4, z4, z8, 0);
}

// ---------------------------------------------------------------------------
// 0) f32 -> f16 conversion
// ---------------------------------------------------------------------------
__global__ void cvt_f16_kernel(const float* __restrict__ src,
                               _Float16* __restrict__ dst, int n) {
  int i = blockIdx.x * blockDim.x + threadIdx.x;
  if (i < n) dst[i] = (_Float16)src[i];
}

// ---------------------------------------------------------------------------
// 1) QKV GEMM: [4096 x 768] x [768 x 2304] (+bias). 32x32 macro-tile per wave
//    (2x2 WMMA tiles, each fragment reused twice). Epilogue scatters:
//    q,k -> [B,H,L,D] f16 ; v -> [B,H,D,L] f16 (transposed for P@V stage)
// ---------------------------------------------------------------------------
__device__ __forceinline__ void store_qkv_tile(const v8f& c, int mbase, int col,
                                               const float* __restrict__ bqkv,
                                               _Float16* __restrict__ qh,
                                               _Float16* __restrict__ kh,
                                               _Float16* __restrict__ vt, int lane) {
  const int t  = col / C_;           // 0=q 1=k 2=v
  const int cc = col - t * C_;
  const int h  = cc >> 6;
  const int d  = cc & 63;
  const float bias = bqkv[col];
#pragma unroll
  for (int r = 0; r < 8; ++r) {
    const int row = mbase + r + ((lane >> 4) * 8);
    const int b_  = row >> 11;       // /L_
    const int l   = row & (L_ - 1);
    const int bh  = b_ * H_ + h;
    const _Float16 v = (_Float16)(c[r] + bias);
    if (t == 0)      qh[((size_t)bh * L_ + l) * D_ + d] = v;
    else if (t == 1) kh[((size_t)bh * L_ + l) * D_ + d] = v;
    else             vt[((size_t)bh * D_ + d) * L_ + l] = v;
  }
}

__global__ void __launch_bounds__(256)
gemm_qkv_kernel(const _Float16* __restrict__ xh, const _Float16* __restrict__ wq,
                const float* __restrict__ bqkv, _Float16* __restrict__ qh,
                _Float16* __restrict__ kh, _Float16* __restrict__ vt) {
  const int lane = threadIdx.x & 31;
  const int wave = threadIdx.x >> 5;
  const int tile = blockIdx.x * 8 + wave;        // 9216 macro-tiles
  const int m0 = (tile / (NQKV / 32)) * 32;
  const int n0 = (tile % (NQKV / 32)) * 32;

  v8f c00 = {}, c01 = {}, c10 = {}, c11 = {};
#pragma unroll 2
  for (int k0 = 0; k0 < C_; k0 += 32) {
    const v16h a0 = load_frag_a(xh, C_, m0,      k0, lane);
    const v16h a1 = load_frag_a(xh, C_, m0 + 16, k0, lane);
    const v16h b0 = load_frag_b(wq, C_, n0,      k0, lane);
    const v16h b1 = load_frag_b(wq, C_, n0 + 16, k0, lane);
    c00 = wmma_f16(a0, b0, c00);
    c01 = wmma_f16(a0, b1, c01);
    c10 = wmma_f16(a1, b0, c10);
    c11 = wmma_f16(a1, b1, c11);
  }
  const int coln = lane & 15;
  store_qkv_tile(c00, m0,      n0 + coln,      bqkv, qh, kh, vt, lane);
  store_qkv_tile(c01, m0,      n0 + 16 + coln, bqkv, qh, kh, vt, lane);
  store_qkv_tile(c10, m0 + 16, n0 + coln,      bqkv, qh, kh, vt, lane);
  store_qkv_tile(c11, m0 + 16, n0 + 16 + coln, bqkv, qh, kh, vt, lane);
}

// ---------------------------------------------------------------------------
// 2) l2-normalize q (and scale by exp(min(logscale, ln100))) and k
// ---------------------------------------------------------------------------
__global__ void normqk_kernel(_Float16* __restrict__ qh, _Float16* __restrict__ kh,
                              const float* __restrict__ sml) {
  const int idx = blockIdx.x * blockDim.x + threadIdx.x;   // [0, B*H*L)
  const int bh = idx / L_;
  const int h  = bh % H_;
  _Float16* q = qh + (size_t)idx * D_;
  _Float16* k = kh + (size_t)idx * D_;
  float sq = 0.f, sk = 0.f;
#pragma unroll 8
  for (int d = 0; d < D_; ++d) {
    float fq = (float)q[d], fk = (float)k[d];
    sq += fq * fq; sk += fk * fk;
  }
  const float smul = __expf(fminf(sml[h], MAXLOGSCALE));
  const float rq = smul / fmaxf(sqrtf(sq), 1e-12f);
  const float rk = 1.0f / fmaxf(sqrtf(sk), 1e-12f);
#pragma unroll 8
  for (int d = 0; d < D_; ++d) {
    q[d] = (_Float16)((float)q[d] * rq);
    k[d] = (_Float16)((float)k[d] * rk);
  }
}

// ---------------------------------------------------------------------------
// 3) Flash attention: one wave per 16-row Q tile, 32 keys/step, online
//    softmax; K/V tiles staged into LDS by the Tensor Data Mover.
// ---------------------------------------------------------------------------
__global__ void __launch_bounds__(256)
flash_kernel(const _Float16* __restrict__ qh, const _Float16* __restrict__ kh,
             const _Float16* __restrict__ vt, const float* __restrict__ bias,
             _Float16* __restrict__ ao) {
  __shared__ _Float16 ldsP[8 * 16 * 32];   //  8 KB: P re-layout buffer
  __shared__ _Float16 ldsK[8 * 32 * 64];   // 32 KB: K tile  [32 keys x 64 d]
  __shared__ _Float16 ldsV[8 * 64 * 32];   // 32 KB: V tile  [64 d x 32 keys]
  const int lane = threadIdx.x & 31;
  const int wave = threadIdx.x >> 5;
  const int tile = blockIdx.x * 8 + wave;      // 3072 tiles
  const int qtile = tile & 127;                // L_/16 = 128
  const int bh   = tile >> 7;
  const int b_   = bh / H_;
  const int h    = bh % H_;
  const int q0   = qtile * 16;

  const _Float16* qb = qh + (size_t)bh * L_ * D_;
  const _Float16* kb = kh + (size_t)bh * L_ * D_;
  const _Float16* vb = vt + (size_t)bh * D_ * L_;
  _Float16* lp = ldsP + wave * (16 * 32);
  _Float16* lk = ldsK + wave * (32 * 64);
  _Float16* lv = ldsV + wave * (64 * 32);

  const v16h a0 = load_frag_a(qb, D_, q0, 0, lane);
  const v16h a1 = load_frag_a(qb, D_, q0, 32, lane);

  v8f o0 = {}, o1 = {}, o2 = {}, o3 = {};
  float m[8], l[8];
#pragma unroll
  for (int r = 0; r < 8; ++r) { m[r] = -1e30f; l[r] = 0.f; }

#pragma unroll 1
  for (int kk0 = 0; kk0 < L_; kk0 += 32) {
    __builtin_prefetch(kb + (size_t)(kk0 + 32) * D_, 0, 1);
    __builtin_prefetch(vb + kk0 + 32, 0, 1);

    // ---- TDM: stage K [32 x 64] (contiguous rows) and V [64 x 32] (strided)
    tdm_load_tile((unsigned int)(size_t)lk, kb + (size_t)kk0 * D_, 64, 32, D_);
    tdm_load_tile((unsigned int)(size_t)lv, vb + kk0, 32, 64, L_);
    __builtin_amdgcn_s_wait_tensorcnt(0);

    // ---- S = Q K^T (two 16-col tiles) from LDS, + attn_bias ----
    v8f s0 = {}, s1 = {};
    {
      v16h kb0 = load_frag_b(lk, D_, 0, 0, lane);
      v16h kb1 = load_frag_b(lk, D_, 0, 32, lane);
      s0 = wmma_f16(a0, kb0, s0);
      s0 = wmma_f16(a1, kb1, s0);
      v16h kc0 = load_frag_b(lk, D_, 16, 0, lane);
      v16h kc1 = load_frag_b(lk, D_, 16, 32, lane);
      s1 = wmma_f16(a0, kc0, s1);
      s1 = wmma_f16(a1, kc1, s1);
    }
    const int coln = lane & 15;
#pragma unroll
    for (int r = 0; r < 8; ++r) {
      const int row = q0 + r + ((lane >> 4) * 8);
      s0[r] += bias[(size_t)row * L_ + kk0 + coln];
      s1[r] += bias[(size_t)row * L_ + kk0 + 16 + coln];
    }

    // ---- online softmax (row reductions across 16-lane groups) ----
    float e0[8], e1[8], corr[8];
#pragma unroll
    for (int r = 0; r < 8; ++r) {
      float v = fmaxf(s0[r], s1[r]);
      v = fmaxf(v, __shfl_xor(v, 1, 32));
      v = fmaxf(v, __shfl_xor(v, 2, 32));
      v = fmaxf(v, __shfl_xor(v, 4, 32));
      v = fmaxf(v, __shfl_xor(v, 8, 32));
      const float mnew = fmaxf(m[r], v);
      corr[r] = __expf(m[r] - mnew);
      e0[r] = __expf(s0[r] - mnew);
      e1[r] = __expf(s1[r] - mnew);
      float ps = e0[r] + e1[r];
      ps += __shfl_xor(ps, 1, 32);
      ps += __shfl_xor(ps, 2, 32);
      ps += __shfl_xor(ps, 4, 32);
      ps += __shfl_xor(ps, 8, 32);
      l[r] = l[r] * corr[r] + ps;
      m[r] = mnew;
    }
#pragma unroll
    for (int r = 0; r < 8; ++r) {
      o0[r] *= corr[r]; o1[r] *= corr[r]; o2[r] *= corr[r]; o3[r] *= corr[r];
    }

    // ---- re-layout P (C-frag) -> A-frag via LDS (same-wave DS is in-order) --
#pragma unroll
    for (int r = 0; r < 8; ++r) {
      const int row = r + ((lane >> 4) * 8);
      lp[row * 32 + coln]      = (_Float16)e0[r];
      lp[row * 32 + coln + 16] = (_Float16)e1[r];
    }
    asm volatile("s_wait_dscnt 0x0" ::: "memory");
    const v16h pa = load_frag_a(lp, 32, 0, 0, lane);

    // ---- O += P V (V tile from LDS, ld = 32) ----
    o0 = wmma_f16(pa, load_frag_b(lv, 32, 0,  0, lane), o0);
    o1 = wmma_f16(pa, load_frag_b(lv, 32, 16, 0, lane), o1);
    o2 = wmma_f16(pa, load_frag_b(lv, 32, 32, 0, lane), o2);
    o3 = wmma_f16(pa, load_frag_b(lv, 32, 48, 0, lane), o3);
  }

  // ---- finalize: O /= l, store as f16 [B, L, C] with c = h*64 + d ----
  const int coln = lane & 15;
#pragma unroll
  for (int r = 0; r < 8; ++r) {
    const float inv = 1.0f / l[r];
    const int row = q0 + r + ((lane >> 4) * 8);
    _Float16* dst = ao + ((size_t)b_ * L_ + row) * C_ + h * D_;
    dst[0 * 16 + coln]  = (_Float16)(o0[r] * inv);
    dst[1 * 16 + coln]  = (_Float16)(o1[r] * inv);
    dst[2 * 16 + coln]  = (_Float16)(o2[r] * inv);
    dst[3 * 16 + coln]  = (_Float16)(o3[r] * inv);
  }
}

// ---------------------------------------------------------------------------
// 4) Output projection: [4096 x 768] x [768 x 768] (+bias), 32x32 per wave
// ---------------------------------------------------------------------------
__device__ __forceinline__ void store_out_tile(const v8f& c, int mbase, int col,
                                               const float* __restrict__ bp,
                                               float* __restrict__ out, int lane) {
  const float bias = bp[col];
#pragma unroll
  for (int r = 0; r < 8; ++r) {
    const int row = mbase + r + ((lane >> 4) * 8);
    out[(size_t)row * C_ + col] = c[r] + bias;
  }
}

__global__ void __launch_bounds__(256)
gemm_proj_kernel(const _Float16* __restrict__ ah, const _Float16* __restrict__ wp,
                 const float* __restrict__ bp, float* __restrict__ out) {
  const int lane = threadIdx.x & 31;
  const int wave = threadIdx.x >> 5;
  const int tile = blockIdx.x * 8 + wave;      // 3072 macro-tiles
  const int m0 = (tile / (C_ / 32)) * 32;
  const int n0 = (tile % (C_ / 32)) * 32;

  v8f c00 = {}, c01 = {}, c10 = {}, c11 = {};
#pragma unroll 2
  for (int k0 = 0; k0 < C_; k0 += 32) {
    const v16h a0 = load_frag_a(ah, C_, m0,      k0, lane);
    const v16h a1 = load_frag_a(ah, C_, m0 + 16, k0, lane);
    const v16h b0 = load_frag_b(wp, C_, n0,      k0, lane);
    const v16h b1 = load_frag_b(wp, C_, n0 + 16, k0, lane);
    c00 = wmma_f16(a0, b0, c00);
    c01 = wmma_f16(a0, b1, c01);
    c10 = wmma_f16(a1, b0, c10);
    c11 = wmma_f16(a1, b1, c11);
  }
  const int coln = lane & 15;
  store_out_tile(c00, m0,      n0 + coln,      bp, out, lane);
  store_out_tile(c01, m0,      n0 + 16 + coln, bp, out, lane);
  store_out_tile(c10, m0 + 16, n0 + coln,      bp, out, lane);
  store_out_tile(c11, m0 + 16, n0 + 16 + coln, bp, out, lane);
}

// ---------------------------------------------------------------------------
// launch
// ---------------------------------------------------------------------------
extern "C" void kernel_launch(void* const* d_in, const int* in_sizes, int n_in,
                              void* d_out, int out_size, void* d_ws, size_t ws_size,
                              hipStream_t stream) {
  const float* x      = (const float*)d_in[0];
  const float* abias  = (const float*)d_in[1];
  const float* W_qkv  = (const float*)d_in[2];
  const float* b_qkv  = (const float*)d_in[3];
  const float* sml    = (const float*)d_in[4];
  const float* W_proj = (const float*)d_in[5];
  const float* b_proj = (const float*)d_in[6];
  float* out = (float*)d_out;

  char* ws = (char*)d_ws;
  _Float16* xh  = (_Float16*)(ws + 0);                        // 4096*768
  _Float16* wqh = (_Float16*)(ws + 6291456);                  // 2304*768
  _Float16* wph = (_Float16*)(ws + 9830400);                  // 768*768
  _Float16* qh  = (_Float16*)(ws + 11010048);                 // B*H*L*D
  _Float16* kh  = (_Float16*)(ws + 17301504);                 // B*H*L*D
  _Float16* vt  = (_Float16*)(ws + 23592960);                 // B*H*D*L
  _Float16* ao  = (_Float16*)(ws + 29884416);                 // 4096*768

  cvt_f16_kernel<<<(M_ * C_) / 256, 256, 0, stream>>>(x, xh, M_ * C_);
  cvt_f16_kernel<<<(NQKV * C_) / 256, 256, 0, stream>>>(W_qkv, wqh, NQKV * C_);
  cvt_f16_kernel<<<(C_ * C_) / 256, 256, 0, stream>>>(W_proj, wph, C_ * C_);

  gemm_qkv_kernel<<<(M_ / 32) * (NQKV / 32) / 8, 256, 0, stream>>>(
      xh, wqh, b_qkv, qh, kh, vt);

  normqk_kernel<<<(B_ * H_ * L_) / 256, 256, 0, stream>>>(qh, kh, sml);

  flash_kernel<<<(B_ * H_ * (L_ / 16)) / 8, 256, 0, stream>>>(
      qh, kh, vt, abias, ao);

  gemm_proj_kernel<<<(M_ / 32) * (C_ / 32) / 8, 256, 0, stream>>>(
      ao, wph, b_proj, out);
}
